// RcMoE_Rep_Layer_58712202936378
// MI455X (gfx1250) — matrix operand order repaired and checked by reading, not compiled, not run on hardware
//
#include <hip/hip_runtime.h>

// ---------------------------------------------------------------------------
// Types (plain ext-vectors so they are trivially constructible in unions)
// ---------------------------------------------------------------------------
typedef __bf16 bf16;
typedef __bf16 v16bf __attribute__((ext_vector_type(16)));
typedef float  v8f   __attribute__((ext_vector_type(8)));
typedef float  f32x4 __attribute__((ext_vector_type(4)));
typedef unsigned int u32;
typedef u32    u32x4 __attribute__((ext_vector_type(4)));
typedef int    s32x4 __attribute__((ext_vector_type(4)));

// Generic (gcc-style) vector type: matches the builtin's expected pointee type
typedef int v4i __attribute__((vector_size(16)));
typedef __attribute__((address_space(1))) v4i* gptr_v4i;   // pointer to AS1 v4i
typedef __attribute__((address_space(3))) v4i* lptr_v4i;   // pointer to AS3 v4i

union Frag16 { v16bf v; u32x4 q[2]; };

// Problem constants
#define NTOK 4096   // B*L
#define HDIM 2048
#define EEXP 8
#define DFF  8192
#define NF4_BLK 512

// GEMM tiling
#define TM   128
#define TN   128
#define TK   32
#define LDSK 40     // padded bf16 stride: 80 bytes, multiple of 16 -> b128 LDS ops

// ---------------------------------------------------------------------------
// CDNA5 async global->LDS copy (bypasses VGPRs, tracked by ASYNCcnt).
// Falls back to a synchronous reg-staged copy if the builtin is absent.
// ---------------------------------------------------------------------------
#if __has_builtin(__builtin_amdgcn_global_load_async_to_lds_b128)
#define ASYNC_TILE 1
#else
#define ASYNC_TILE 0
#endif

__device__ __forceinline__ void copy16_g2l(bf16* ldst, const bf16* gsrc)
{
#if ASYNC_TILE
    __builtin_amdgcn_global_load_async_to_lds_b128(
        (gptr_v4i)(uintptr_t)gsrc,
        (lptr_v4i)(unsigned)(uintptr_t)ldst,
        0, 0);
#else
    *(u32x4*)ldst = *(const u32x4*)gsrc;
#endif
}

__device__ __forceinline__ void wait_async_zero()
{
#if ASYNC_TILE
#if __has_builtin(__builtin_amdgcn_s_wait_asynccnt)
    __builtin_amdgcn_s_wait_asynccnt(0);
#else
    asm volatile("s_wait_asynccnt 0x0" ::: "memory");
#endif
#endif
}

__device__ __forceinline__ float gelu_tanh(float v) {
    const float k = 0.7978845608028654f;   // sqrt(2/pi)
    float t = tanhf(k * (v + 0.044715f * v * v * v));
    return 0.5f * v * (1.0f + t);
}

// ---------------------------------------------------------------------------
// Issue one 128x32 bf16 tile copy: global (bf16, row-major, ld=K) -> padded LDS
// 256 threads, 8 elements (16 B) per thread per iteration, 2 iterations
// ---------------------------------------------------------------------------
__device__ __forceinline__ void issue_tile128x32(
    bf16* __restrict__ dst, const bf16* __restrict__ src,
    int ld, int row0, int col0, int tid)
{
    const int r = tid >> 2;          // 0..63
    const int c = (tid & 3) << 3;    // 0,8,16,24
#pragma unroll
    for (int it = 0; it < 2; ++it) {
        const int row = r + (it << 6);
        copy16_g2l(&dst[row * LDSK + c],
                   src + (size_t)(row0 + row) * ld + col0 + c);
    }
}

// ---------------------------------------------------------------------------
// WMMA GEMM: C[M,N] = act( A[M,K] @ Bt[N,K]^T ), A/Bt bf16 row-major
// Block = 256 threads (8 waves), tile 128x128, K-step 32, double-buffered LDS.
// Pipeline: wait_async(0) -> barrier -> issue tile k+1 -> compute tile k.
// Wave (wm,wn) owns a 32x64 sub-tile = 2x4 WMMA 16x16 accumulators.
// ---------------------------------------------------------------------------
template <bool GELU_BF16_OUT>
__global__ __launch_bounds__(256)
void wmma_gemm_kernel(const bf16* __restrict__ A, const bf16* __restrict__ Bt,
                      void* __restrict__ C, int M, int K, int N)
{
    __shared__ bf16 As[2][TM * LDSK];
    __shared__ bf16 Bs[2][TN * LDSK];

    const int tid  = threadIdx.x;
    const int lane = tid & 31;
    const int half = lane >> 4;      // 0/1: lane group (K-half selector, ISA layout)
    const int l16  = lane & 15;      // row (A) / col (B,C)
    const int wid  = tid >> 5;
    const int wm   = (wid & 3) << 5; // 4 wave-rows * 32
    const int wn   = (wid >> 2) << 6;// 2 wave-cols * 64
    const int bm   = blockIdx.y << 7;
    const int bn   = blockIdx.x << 7;

    v8f acc[2][4] = {};
    const int nK = K / TK;

    // Prologue: start tile 0 into buffer 0
    issue_tile128x32(As[0], A, K, bm, 0, tid);
    issue_tile128x32(Bs[0], Bt, K, bn, 0, tid);

    for (int kt = 0; kt < nK; ++kt) {
        const int cur = kt & 1;

        wait_async_zero();   // my tile-kt copies have landed in LDS
        __syncthreads();     // all waves' copies landed; prev compute finished

        if (kt + 1 < nK) {   // overlap tile k+1 DMA with tile k WMMA
            issue_tile128x32(As[cur ^ 1], A, K, bm, (kt + 1) * TK, tid);
            issue_tile128x32(Bs[cur ^ 1], Bt, K, bn, (kt + 1) * TK, tid);
        }

        // A fragment: lane holds row m=l16; K = half*8..+7 then 16+half*8..+7
        Frag16 af[2];
#pragma unroll
        for (int mi = 0; mi < 2; ++mi) {
            const bf16* p = &As[cur][(wm + (mi << 4) + l16) * LDSK];
            af[mi].q[0] = *(const u32x4*)(p + (half << 3));
            af[mi].q[1] = *(const u32x4*)(p + 16 + (half << 3));
        }
        // B fragment: lane holds col n=l16; K = half*8..+7 then 16+half*8..+7 (N-major LDS tile)
        Frag16 bfr[4];
#pragma unroll
        for (int ni = 0; ni < 4; ++ni) {
            const bf16* p = &Bs[cur][(wn + (ni << 4) + l16) * LDSK];
            bfr[ni].q[0] = *(const u32x4*)(p + (half << 3));
            bfr[ni].q[1] = *(const u32x4*)(p + 16 + (half << 3));
        }
#pragma unroll
        for (int mi = 0; mi < 2; ++mi)
#pragma unroll
            for (int ni = 0; ni < 4; ++ni)
                acc[mi][ni] = __builtin_amdgcn_wmma_f32_16x16x32_bf16(
                    false, af[mi].v, false, bfr[ni].v,
                    (short)0, acc[mi][ni], false, false);

        __syncthreads();     // all waves done reading buf[cur] before next overwrite
    }

    // Epilogue: C/D layout -> lane covers col l16, rows r + half*8
#pragma unroll
    for (int mi = 0; mi < 2; ++mi)
#pragma unroll
        for (int ni = 0; ni < 4; ++ni) {
            const int row0 = bm + wm + (mi << 4) + (half << 3);
            const int col  = bn + wn + (ni << 4) + l16;
#pragma unroll
            for (int r = 0; r < 8; ++r) {
                float v = acc[mi][ni][r];
                if constexpr (GELU_BF16_OUT) {
                    ((bf16*)C)[(size_t)(row0 + r) * N + col] = (bf16)gelu_tanh(v);
                } else {
                    ((float*)C)[(size_t)(row0 + r) * N + col] = v;
                }
            }
        }
}

// ---------------------------------------------------------------------------
// fp32 -> bf16 straight conversion (x -> xb), 8 elements/thread
// ---------------------------------------------------------------------------
__global__ __launch_bounds__(256)
void convert_f32_to_bf16(const float* __restrict__ src, bf16* __restrict__ dst)
{
    const size_t i = ((size_t)blockIdx.x * 256 + threadIdx.x) * 8;
    f32x4 f0 = *(const f32x4*)(src + i);
    f32x4 f1 = *(const f32x4*)(src + i + 4);
    union { bf16 h[8]; u32x4 q; } u;
    u.h[0] = (bf16)f0.x; u.h[1] = (bf16)f0.y; u.h[2] = (bf16)f0.z; u.h[3] = (bf16)f0.w;
    u.h[4] = (bf16)f1.x; u.h[5] = (bf16)f1.y; u.h[6] = (bf16)f1.z; u.h[7] = (bf16)f1.w;
    *(u32x4*)(dst + i) = u.q;
}

// ---------------------------------------------------------------------------
// fp32 [R,C] -> bf16 transposed [C,R]  (LDS-tiled 32x32)
// ---------------------------------------------------------------------------
__global__ __launch_bounds__(256)
void transpose_f32_to_bf16(const float* __restrict__ src, bf16* __restrict__ dst,
                           int R, int C)
{
    __shared__ float tile[32][33];
    const int bx = blockIdx.x << 5;   // col of src
    const int by = blockIdx.y << 5;   // row of src
    const int tx = threadIdx.x;       // 0..31
    const int ty = threadIdx.y;       // 0..7
#pragma unroll
    for (int j = 0; j < 4; ++j)
        tile[ty + (j << 3)][tx] = src[(size_t)(by + ty + (j << 3)) * C + bx + tx];
    __syncthreads();
#pragma unroll
    for (int j = 0; j < 4; ++j)
        dst[(size_t)(bx + ty + (j << 3)) * R + by + tx] = (bf16)tile[tx][ty + (j << 3)];
}

// ---------------------------------------------------------------------------
// Router: logits = x[n,:] @ rw[:,E], softmax, round through fp16
// One block per token row.
// ---------------------------------------------------------------------------
__global__ __launch_bounds__(256)
void router_softmax_kernel(const float* __restrict__ x, const float* __restrict__ rw,
                           float* __restrict__ probs)
{
    const int n   = blockIdx.x;
    const int tid = threadIdx.x;
    const float* xr = x + (size_t)n * HDIM;

    float acc[EEXP] = {};
    for (int h = tid; h < HDIM; h += 256) {
        const float xv = xr[h];
        const float* w = rw + (size_t)h * EEXP;
#pragma unroll
        for (int e = 0; e < EEXP; ++e) acc[e] += xv * w[e];
    }
#pragma unroll
    for (int off = 16; off > 0; off >>= 1)
#pragma unroll
        for (int e = 0; e < EEXP; ++e)
            acc[e] += __shfl_down(acc[e], off, 32);

    __shared__ float part[8][EEXP];
    const int lane = tid & 31, wid = tid >> 5;
    if (lane == 0)
#pragma unroll
        for (int e = 0; e < EEXP; ++e) part[wid][e] = acc[e];
    __syncthreads();

    if (tid == 0) {
        float lg[EEXP];
#pragma unroll
        for (int e = 0; e < EEXP; ++e) {
            float s = 0.f;
#pragma unroll
            for (int w = 0; w < 8; ++w) s += part[w][e];
            lg[e] = s;
        }
        float mx = lg[0];
#pragma unroll
        for (int e = 1; e < EEXP; ++e) mx = fmaxf(mx, lg[e]);
        float sum = 0.f;
#pragma unroll
        for (int e = 0; e < EEXP; ++e) { lg[e] = __expf(lg[e] - mx); sum += lg[e]; }
        const float inv = 1.0f / sum;
#pragma unroll
        for (int e = 0; e < EEXP; ++e) {
            float p = lg[e] * inv;
            p = (float)(_Float16)p;          // fp16 round-trip as in reference
            probs[(size_t)n * EEXP + e] = p;
        }
    }
}

// ---------------------------------------------------------------------------
// NF4 dequant + expert mix + residual add into out (out already holds MLP)
// One block per token row; thread t handles h = 8t..8t+7 (stays in one 512-block)
// ---------------------------------------------------------------------------
__global__ __launch_bounds__(256)
void moe_dequant_mix_kernel(const int* __restrict__ nf4,
                            const float* __restrict__ mean,
                            const float* __restrict__ stdv,
                            const float* __restrict__ cbg,
                            const float* __restrict__ probs,
                            float* __restrict__ out)
{
    const int n   = blockIdx.x;
    const int tid = threadIdx.x;
    const int h0  = tid << 3;          // 8 h-values per thread

    __shared__ float cb[16];
    __shared__ float p[EEXP];
    if (tid < 16) cb[tid] = cbg[tid];
    if (tid < EEXP) p[tid] = probs[(size_t)n * EEXP + tid];
    __syncthreads();

    const int blk = h0 >> 9;           // h0 / 512
    float moe[8] = {};
#pragma unroll
    for (int e = 0; e < EEXP; ++e) {
        const float pe = p[e];
        const size_t ne = (size_t)(n * EEXP + e);
        const s32x4 packed = *(const s32x4*)(nf4 + ne * (HDIM / 2) + (h0 >> 1));
        const float sd = stdv[ne * (HDIM / NF4_BLK) + blk];
        const float mn = mean[ne * (HDIM / NF4_BLK) + blk];
        int v[4] = { packed.x, packed.y, packed.z, packed.w };
#pragma unroll
        for (int i = 0; i < 4; ++i) {
            const int b = v[i] & 255;
            moe[2 * i]     += pe * (cb[b & 15]        * sd + mn);   // low nibble -> even h
            moe[2 * i + 1] += pe * (cb[(b >> 4) & 15] * sd + mn);   // high nibble -> odd h
        }
    }

    float* o = out + ((size_t)n << 11) + h0;
    f32x4 a0 = *(f32x4*)o;
    f32x4 a1 = *(f32x4*)(o + 4);
    a0.x += moe[0]; a0.y += moe[1]; a0.z += moe[2]; a0.w += moe[3];
    a1.x += moe[4]; a1.y += moe[5]; a1.z += moe[6]; a1.w += moe[7];
    *(f32x4*)o = a0;
    *(f32x4*)(o + 4) = a1;
}

// ---------------------------------------------------------------------------
// Launch
// inputs: 0=x 1=router_w 2=nf4 3=mean 4=std 5=codebook 6=w1 7=w2
// ws: w1t bf16 [DFF,H] | w2t bf16 [H,DFF] | act bf16 [N,DFF] | xb bf16 [N,H] | probs
// ---------------------------------------------------------------------------
extern "C" void kernel_launch(void* const* d_in, const int* in_sizes, int n_in,
                              void* d_out, int out_size, void* d_ws, size_t ws_size,
                              hipStream_t stream)
{
    (void)in_sizes; (void)n_in; (void)out_size; (void)ws_size;
    const float* x   = (const float*)d_in[0];
    const float* rw  = (const float*)d_in[1];
    const int*   nf4 = (const int*)d_in[2];
    const float* mean= (const float*)d_in[3];
    const float* stdv= (const float*)d_in[4];
    const float* cbg = (const float*)d_in[5];
    const float* w1  = (const float*)d_in[6];
    const float* w2  = (const float*)d_in[7];

    char* ws = (char*)d_ws;
    bf16*  w1t   = (bf16*)(ws);                                        // 32 MiB
    bf16*  w2t   = (bf16*)(ws + (size_t)DFF * HDIM * 2);               // 32 MiB
    bf16*  act   = (bf16*)(ws + (size_t)DFF * HDIM * 4);               // 64 MiB
    bf16*  xb    = (bf16*)(ws + (size_t)DFF * HDIM * 4
                              + (size_t)NTOK * DFF * 2);               // 16 MiB
    float* probs = (float*)(ws + (size_t)DFF * HDIM * 4
                               + (size_t)NTOK * DFF * 2
                               + (size_t)NTOK * HDIM * 2);
    float* outp  = (float*)d_out;

    dim3 tb(32, 8);
    // w1 [H,DFF] -> w1t [DFF,H]
    transpose_f32_to_bf16<<<dim3(DFF / 32, HDIM / 32), tb, 0, stream>>>(w1, w1t, HDIM, DFF);
    // w2 [DFF,H] -> w2t [H,DFF]
    transpose_f32_to_bf16<<<dim3(HDIM / 32, DFF / 32), tb, 0, stream>>>(w2, w2t, DFF, HDIM);
    // x -> xb (bf16)
    convert_f32_to_bf16<<<(NTOK * HDIM) / (256 * 8), 256, 0, stream>>>(x, xb);

    router_softmax_kernel<<<NTOK, 256, 0, stream>>>(x, rw, probs);

    // GEMM1: act = gelu(xb @ w1)  (bf16 out)
    wmma_gemm_kernel<true><<<dim3(DFF / TN, NTOK / TM), 256, 0, stream>>>(
        xb, w1t, (void*)act, NTOK, HDIM, DFF);

    // GEMM2: out = act @ w2  (fp32 straight to d_out)
    wmma_gemm_kernel<false><<<dim3(HDIM / TN, NTOK / TM), 256, 0, stream>>>(
        act, w2t, (void*)outp, NTOK, DFF, HDIM);

    // out += sum_e p_e * dequant_nf4(e)
    moe_dequant_mix_kernel<<<NTOK, 256, 0, stream>>>(nf4, mean, stdv, cbg, probs, outp);
}
